// KimiDeltaAttention_3418793968186
// MI455X (gfx1250) — matrix operand order repaired and checked by reading, not compiled, not run on hardware
//
#include <hip/hip_runtime.h>
#include <hip/hip_bf16.h>

// ---------------------------------------------------------------------------
// Kimi Delta Attention for MI455X (gfx1250, wave32, WMMA).
// - All dense GEMMs: v_wmma_f32_16x16x32_bf16 (bf16 in, f32 accum).
// - The delta-rule recurrence uses the CHUNKED formulation (C=64) so the
//   intra/inter-chunk work is WMMA matmuls; the 128x128 state lives in the
//   WMMA accumulators of one workgroup per (b,h) across all 64 chunks.
// ---------------------------------------------------------------------------

typedef __attribute__((ext_vector_type(16))) __bf16 v16bf;
typedef __attribute__((ext_vector_type(8)))  float  v8f;

__device__ __forceinline__ unsigned short f2bf(float f) {
  union { float f; unsigned u; } x; x.f = f;
  unsigned r = x.u + 0x7FFFu + ((x.u >> 16) & 1u);   // round-to-nearest-even
  return (unsigned short)(r >> 16);
}
__device__ __forceinline__ float bf2f(unsigned short h) {
  union { unsigned u; float f; } x; x.u = ((unsigned)h) << 16; return x.f;
}

#define WMMA_BF16(a, b, c) __builtin_amdgcn_wmma_f32_16x16x32_bf16( \
    false, (a), false, (b), (short)0, (c), false, false)

// A-fragment (16xK tile slice, ISA layout: elems 0..7 => K=k0+8*half..,
// elems 8..15 => K=k0+16+8*half..) from a row-major bf16 LDS/array.
__device__ __forceinline__ v16bf ldsA(const unsigned short* p0, int ld,
                                      int row, int k0, int l16, int lhalf) {
  const unsigned short* p = p0 + (row + l16) * ld + k0 + 8 * lhalf;
  v16bf a;
  *((uint4*)&a)     = *(const uint4*)p;
  *((uint4*)&a + 1) = *(const uint4*)(p + 16);
  return a;
}
// B-fragment from Bt (=B^T) row-major, same K layout as the A-fragment.
__device__ __forceinline__ v16bf ldsB(const unsigned short* p0, int ld,
                                      int row, int k0, int l16, int lhalf) {
  const unsigned short* p = p0 + (row + l16) * ld + k0 + 8 * lhalf;
  v16bf b;
  *((uint4*)&b)     = *(const uint4*)p;
  *((uint4*)&b + 1) = *(const uint4*)(p + 16);
  return b;
}

union Pack8 { unsigned short us[8]; uint4 v; };

// ------------------------- elementwise converts ----------------------------
__global__ void f32_to_bf16_kernel(const float* __restrict__ in,
                                   unsigned short* __restrict__ out, size_t n) {
  size_t i = (size_t)blockIdx.x * blockDim.x + threadIdx.x;
  if (i < n) out[i] = f2bf(in[i]);
}

// Wt[n][k] = bf16(W[k][n]); W is [K][N] row-major.
__global__ void transpose_bf16_kernel(const float* __restrict__ W,
                                      unsigned short* __restrict__ Wt,
                                      int K, int N) {
  size_t i = (size_t)blockIdx.x * blockDim.x + threadIdx.x;
  size_t total = (size_t)K * N;
  if (i >= total) return;
  int k = (int)(i % K);
  int n = (int)(i / K);
  Wt[(size_t)n * K + k] = f2bf(W[(size_t)k * N + n]);
}

// ------------------------- WMMA tiled GEMM ---------------------------------
// C[M,N] = A[M,K]*B[K,N]. A bf16 row-major, Bt = B^T bf16 [N][K].
constexpr int BM = 128, BN = 128, BK = 32, LDT = 40;

template<int OUT_BF16>
__global__ __launch_bounds__(256)
void wmma_gemm_kernel(const unsigned short* __restrict__ A,
                      const unsigned short* __restrict__ Bt,
                      void* __restrict__ C, int M, int N, int K) {
  (void)M;
  __shared__ __attribute__((aligned(16))) unsigned short As[BM * LDT];
  __shared__ __attribute__((aligned(16))) unsigned short Bs[BN * LDT];

  const int tid   = threadIdx.x;
  const int wave  = tid >> 5;
  const int lane  = tid & 31;
  const int lhalf = lane >> 4;
  const int l16   = lane & 15;
  const int bM = blockIdx.y * BM;
  const int bN = blockIdx.x * BN;
  const int wM = (wave >> 1) * 32;
  const int wN = (wave & 1) * 64;

  v8f acc[2][4];
#pragma unroll
  for (int i = 0; i < 2; ++i)
#pragma unroll
    for (int j = 0; j < 4; ++j)
#pragma unroll
      for (int e = 0; e < 8; ++e) acc[i][j][e] = 0.0f;

  for (int k0 = 0; k0 < K; k0 += BK) {
#pragma unroll
    for (int c = 0; c < 2; ++c) {
      int chunk = tid + c * 256;
      int row   = chunk >> 2;
      int kc    = (chunk & 3) * 8;
      uint4 av = *(const uint4*)(A  + (size_t)(bM + row) * K + k0 + kc);
      *(uint4*)(&As[row * LDT + kc]) = av;
      uint4 bv = *(const uint4*)(Bt + (size_t)(bN + row) * K + k0 + kc);
      *(uint4*)(&Bs[row * LDT + kc]) = bv;
    }
    __syncthreads();
    v16bf af[2];
#pragma unroll
    for (int i = 0; i < 2; ++i) af[i] = ldsA(As, LDT, wM + i * 16, 0, l16, lhalf);
    v16bf bfm[4];
#pragma unroll
    for (int j = 0; j < 4; ++j) bfm[j] = ldsB(Bs, LDT, wN + j * 16, 0, l16, lhalf);
#pragma unroll
    for (int i = 0; i < 2; ++i)
#pragma unroll
      for (int j = 0; j < 4; ++j)
        acc[i][j] = WMMA_BF16(af[i], bfm[j], acc[i][j]);
    __syncthreads();
  }
#pragma unroll
  for (int i = 0; i < 2; ++i)
#pragma unroll
    for (int j = 0; j < 4; ++j) {
      int n = bN + wN + j * 16 + l16;
#pragma unroll
      for (int r = 0; r < 8; ++r) {
        int m = bM + wM + i * 16 + lhalf * 8 + r;
        float val = acc[i][j][r];
        if (OUT_BF16) ((unsigned short*)C)[(size_t)m * N + n] = f2bf(val);
        else          ((float*)C)[(size_t)m * N + n] = val;
      }
    }
}

// ------------------- depthwise causal conv (K=4) + silu --------------------
__global__ void conv_silu_kernel(const unsigned short* __restrict__ xraw,
                                 const float* __restrict__ w,   // [C][4]
                                 unsigned short* __restrict__ y,
                                 int T, int C) {
  size_t idx = (size_t)blockIdx.x * blockDim.x + threadIdx.x;
  int c = (int)(idx % C);
  size_t bt = idx / C;
  int t = (int)(bt % T);
  float acc = 0.f;
#pragma unroll
  for (int j = 0; j < 4; ++j) {
    int tt = t - 3 + j;
    if (tt >= 0)
      acc += bf2f(xraw[(bt - (size_t)(3 - j)) * C + c]) * w[c * 4 + j];
  }
  float s = acc / (1.f + __expf(-acc));   // silu
  y[idx] = f2bf(s);
}

// ----------------- per-head l2 normalization (in place) --------------------
__global__ __launch_bounds__(512)
void l2norm_kernel(unsigned short* __restrict__ qk, float scale) {
  int row  = blockIdx.x;
  int wavh = threadIdx.x >> 5;
  int lane = threadIdx.x & 31;
  size_t base = (size_t)row * 2048 + (size_t)wavh * 128 + (size_t)lane * 4;
  float x0 = bf2f(qk[base + 0]), x1 = bf2f(qk[base + 1]);
  float x2 = bf2f(qk[base + 2]), x3 = bf2f(qk[base + 3]);
  float ss = x0 * x0 + x1 * x1 + x2 * x2 + x3 * x3;
#pragma unroll
  for (int m = 16; m >= 1; m >>= 1) ss += __shfl_xor(ss, m, 32);
  float r = rsqrtf(ss + 1e-6f) * scale;
  qk[base + 0] = f2bf(x0 * r); qk[base + 1] = f2bf(x1 * r);
  qk[base + 2] = f2bf(x2 * r); qk[base + 3] = f2bf(x3 * r);
}

// ------------- g = -exp(A_log[h]) * softplus(g + dt_bias[c]) ---------------
__global__ void kda_gate_kernel(float* __restrict__ g,
                                const float* __restrict__ A_log,
                                const float* __restrict__ dt_bias, size_t n) {
  size_t i = (size_t)blockIdx.x * blockDim.x + threadIdx.x;
  if (i >= n) return;
  int c = (int)(i % 2048);
  int h = c >> 7;
  float xv = g[i] + dt_bias[c];
  float sp = (xv > 20.f) ? xv : log1pf(__expf(xv));
  g[i] = -__expf(A_log[h]) * sp;
}

// --------------------- beta = sigmoid(x @ Wb), N=16 ------------------------
__global__ __launch_bounds__(256)
void beta_kernel(const float* __restrict__ x, const float* __restrict__ Wb,
                 float* __restrict__ beta) {
  __shared__ float red[256];
  int m = blockIdx.x;
  int h = threadIdx.x & 15;
  int chunk = threadIdx.x >> 4;
  float p = 0.f;
  int kbeg = chunk * 128;
  for (int k = kbeg; k < kbeg + 128; ++k)
    p += x[(size_t)m * 2048 + k] * Wb[k * 16 + h];
  red[threadIdx.x] = p;
  __syncthreads();
  if (threadIdx.x < 16) {
    float s = 0.f;
    for (int c = 0; c < 16; ++c) s += red[c * 16 + threadIdx.x];
    beta[(size_t)m * 16 + threadIdx.x] = 1.f / (1.f + __expf(-s));
  }
}

// ---------------- chunked gated delta-rule scan (WMMA) ---------------------
// One workgroup (8 waves) per (b,h); 64 chunks of C=64 steps, D=128.
// State S[128k][128v] lives in WMMA accumulators: wave w owns v-tile column w
// (8 k-tiles = 8 v8f). Per chunk:
//   gc      = within-chunk inclusive cumsum of g            (gl = last row)
//   wq=q*e^gc, wk=k*e^gc, kd=k*e^-gc(clamped), khatT[c][s]=k*e^(gl-gc)
//   u0      = beta*(V - wk @ S0)                            (WMMA)
//   A       = stril(beta * wk @ kd^T), Aq = tril(wq @ kd^T) (WMMA)
//   u       = (I+A)^-1 u0        (per-column forward substitution)
//   O       = wq @ S0 + Aq @ u                              (WMMA)
//   S       = e^gl (.) S + khatT @ u                        (WMMA)
// LDS arena (~145 KB, fine for one WG per 320KB WGP); kd region is reused
// for u^T, gc region is reused for u.
__global__ __launch_bounds__(256)
void kda_chunk_scan_kernel(const unsigned short* __restrict__ q,
                           const unsigned short* __restrict__ k,
                           const unsigned short* __restrict__ v,
                           const float* __restrict__ g,
                           const float* __restrict__ beta,
                           float* __restrict__ o, int T) {
  __shared__ __attribute__((aligned(16))) char smem[148736];
  unsigned short* wq_s  = (unsigned short*)(smem);            // [64][128] bf16
  unsigned short* wk_s  = wq_s + 64 * 128;                    // [64][128]
  unsigned short* kd_s  = wk_s + 64 * 128;                    // [64][128]; later u^T [128][64]
  unsigned short* ut_s  = kd_s;                               // alias
  unsigned short* kht_s = kd_s + 64 * 128;                    // khatT [128][64]
  unsigned short* s0t_s = kht_s + 128 * 64;                   // S0^T [128v][128k]
  float*          gc_s  = (float*)(s0t_s + 128 * 128);        // [64][128] f32
  float*          u_s   = gc_s;                               // alias (after build)
  unsigned short* Ab_s  = (unsigned short*)((char*)gc_s + 64 * 128 * 4); // [64][64]
  unsigned short* Aq_s  = Ab_s + 64 * 64;                     // [64][64]
  float*          gl_s  = (float*)(Aq_s + 64 * 64);           // [128]
  float*          egl_s = gl_s + 128;                         // [128]
  float*          bet_s = egl_s + 128;                        // [64]

  const int h = blockIdx.x, b = blockIdx.y;
  const int tid   = threadIdx.x;
  const int wave  = tid >> 5;
  const int lane  = tid & 31;
  const int lhalf = lane >> 4;
  const int l16   = lane & 15;
  const int tm  = wave >> 1;             // 0..3   (row-tile for 64-row mats)
  const int cnb = (wave & 1) * 4;        // 0/4    (col-tile base, 128-col mats)
  const int tnb = (wave & 1) * 2;        // 0/2    (col-tile base, 64-col mats)

  v8f S[8];                              // state tiles (km=0..7, vn=wave)
#pragma unroll
  for (int km = 0; km < 8; ++km)
#pragma unroll
    for (int e = 0; e < 8; ++e) S[km][e] = 0.f;

  const int NC = T / 64;
  for (int nchunk = 0; nchunk < NC; ++nchunk) {
    const size_t row0 = (size_t)(b * T + nchunk * 64);

    // ---- P0: spill S -> S0^T bf16 (s0t_s[v][k]);  P1a: gc / gl / beta -----
#pragma unroll
    for (int km = 0; km < 8; ++km) {
      Pack8 p;
#pragma unroll
      for (int r = 0; r < 8; ++r) p.us[r] = f2bf(S[km][r]);
      *(uint4*)&s0t_s[(wave * 16 + l16) * 128 + km * 16 + lhalf * 8] = p.v;
    }
    if (tid < 128) {
      float run = 0.f;
      for (int t = 0; t < 64; ++t) {
        run += g[(row0 + t) * 2048 + (size_t)h * 128 + tid];
        gc_s[t * 128 + tid] = run;
      }
      gl_s[tid]  = run;
      egl_s[tid] = __expf(run);
    }
    if (tid < 64) bet_s[tid] = beta[(row0 + tid) * 16 + h];
    __syncthreads();

    // ---- P1b: build wq, wk, kd, khatT --------------------------------------
#pragma unroll
    for (int i = 0; i < 32; ++i) {
      int idx = tid + i * 256;               // 0..8191
      int t = idx >> 7, c = idx & 127;
      size_t go_ = (row0 + t) * 2048 + (size_t)h * 128 + c;
      float gcv = gc_s[idx];
      float e   = __expf(gcv);
      float qv  = bf2f(q[go_]);
      float kv  = bf2f(k[go_]);
      wq_s[idx] = f2bf(qv * e);
      wk_s[idx] = f2bf(kv * e);
      kd_s[idx] = f2bf(kv * __expf(fminf(-gcv, 60.f)));
      kht_s[c * 64 + t] = f2bf(kv * __expf(gl_s[c] - gcv));
    }
    __syncthreads();

    // ---- P2a: u0 = beta*(V - wk @ S0)   [64x128], K=128 --------------------
    {
      v8f U[4];
#pragma unroll
      for (int j = 0; j < 4; ++j)
#pragma unroll
        for (int e = 0; e < 8; ++e) U[j][e] = 0.f;
#pragma unroll
      for (int ks = 0; ks < 4; ++ks) {
        int k0 = ks * 32;
        v16bf a = ldsA(wk_s, 128, tm * 16, k0, l16, lhalf);
#pragma unroll
        for (int j = 0; j < 4; ++j) {
          v16bf bf = ldsB(s0t_s, 128, (cnb + j) * 16, k0, l16, lhalf);
          U[j] = WMMA_BF16(a, bf, U[j]);
        }
      }
      __syncthreads();     // gc_s free -> u_s
#pragma unroll
      for (int j = 0; j < 4; ++j) {
        int c = (cnb + j) * 16 + l16;
#pragma unroll
        for (int r = 0; r < 8; ++r) {
          int t = tm * 16 + lhalf * 8 + r;
          float vv = bf2f(v[(row0 + t) * 2048 + (size_t)h * 128 + c]);
          u_s[t * 128 + c] = bet_s[t] * (vv - U[j][r]);
        }
      }
    }
    __syncthreads();

    // ---- P2b: A = stril(beta*wk@kd^T), Aq = tril(wq@kd^T)  [64x64] ---------
    {
      v8f Aa[2], Aq[2];
#pragma unroll
      for (int j = 0; j < 2; ++j)
#pragma unroll
        for (int e = 0; e < 8; ++e) { Aa[j][e] = 0.f; Aq[j][e] = 0.f; }
#pragma unroll
      for (int ks = 0; ks < 4; ++ks) {
        int k0 = ks * 32;
        v16bf ak = ldsA(wk_s, 128, tm * 16, k0, l16, lhalf);
        v16bf aq = ldsA(wq_s, 128, tm * 16, k0, l16, lhalf);
#pragma unroll
        for (int j = 0; j < 2; ++j) {
          v16bf bf = ldsB(kd_s, 128, (tnb + j) * 16, k0, l16, lhalf);
          Aa[j] = WMMA_BF16(ak, bf, Aa[j]);
          Aq[j] = WMMA_BF16(aq, bf, Aq[j]);
        }
      }
#pragma unroll
      for (int j = 0; j < 2; ++j) {
        int s = (tnb + j) * 16 + l16;
#pragma unroll
        for (int r = 0; r < 8; ++r) {
          int t = tm * 16 + lhalf * 8 + r;
          Ab_s[t * 64 + s] = f2bf((s < t)  ? bet_s[t] * Aa[j][r] : 0.f);
          Aq_s[t * 64 + s] = f2bf((s <= t) ? Aq[j][r] : 0.f);
        }
      }
    }
    __syncthreads();

    // ---- P3: forward substitution u = (I+A)^-1 u0; write u^T bf16 ----------
    if (tid < 128) {
      const int c = tid;
      for (int t = 1; t < 64; ++t) {
        float acc = u_s[t * 128 + c];
        for (int s = 0; s < t; ++s)
          acc -= bf2f(Ab_s[t * 64 + s]) * u_s[s * 128 + c];
        u_s[t * 128 + c] = acc;
      }
#pragma unroll
      for (int sb = 0; sb < 64; sb += 8) {
        Pack8 p;
#pragma unroll
        for (int r = 0; r < 8; ++r) p.us[r] = f2bf(u_s[(sb + r) * 128 + c]);
        *(uint4*)&ut_s[c * 64 + sb] = p.v;    // overwrites kd (dead)
      }
    }
    __syncthreads();

    // ---- P4: O = wq @ S0 + Aq @ u  [64x128] --------------------------------
    {
      v8f O[4];
#pragma unroll
      for (int j = 0; j < 4; ++j)
#pragma unroll
        for (int e = 0; e < 8; ++e) O[j][e] = 0.f;
#pragma unroll
      for (int ks = 0; ks < 4; ++ks) {
        int k0 = ks * 32;
        v16bf a = ldsA(wq_s, 128, tm * 16, k0, l16, lhalf);
#pragma unroll
        for (int j = 0; j < 4; ++j) {
          v16bf bf = ldsB(s0t_s, 128, (cnb + j) * 16, k0, l16, lhalf);
          O[j] = WMMA_BF16(a, bf, O[j]);
        }
      }
#pragma unroll
      for (int ks = 0; ks < 2; ++ks) {
        int k0 = ks * 32;
        v16bf a = ldsA(Aq_s, 64, tm * 16, k0, l16, lhalf);
#pragma unroll
        for (int j = 0; j < 4; ++j) {
          v16bf bf = ldsB(ut_s, 64, (cnb + j) * 16, k0, l16, lhalf);
          O[j] = WMMA_BF16(a, bf, O[j]);
        }
      }
#pragma unroll
      for (int j = 0; j < 4; ++j) {
        int c = (cnb + j) * 16 + l16;
#pragma unroll
        for (int r = 0; r < 8; ++r) {
          int t = tm * 16 + lhalf * 8 + r;
          o[(row0 + t) * 2048 + (size_t)h * 128 + c] = O[j][r];
        }
      }
    }

    // ---- S update: S = e^gl (.) S + khatT @ u  (K=64) ----------------------
#pragma unroll
    for (int km = 0; km < 8; ++km)
#pragma unroll
      for (int r = 0; r < 8; ++r)
        S[km][r] *= egl_s[km * 16 + lhalf * 8 + r];
#pragma unroll
    for (int ks = 0; ks < 2; ++ks) {
      int k0 = ks * 32;
      v16bf bf = ldsB(ut_s, 64, wave * 16, k0, l16, lhalf);
#pragma unroll
      for (int km = 0; km < 8; ++km) {
        v16bf a = ldsA(kht_s, 64, km * 16, k0, l16, lhalf);
        S[km] = WMMA_BF16(a, bf, S[km]);
      }
    }
    __syncthreads();   // protect s0t/ut/kht before next chunk rebuilds them
  }
}

// ------------- o = rmsnorm(o)*w * sigmoid(go)  (per head, D=128) -----------
__global__ __launch_bounds__(512)
void gate_out_kernel(const float* __restrict__ o,
                     const unsigned short* __restrict__ go,
                     const float* __restrict__ w,
                     unsigned short* __restrict__ out) {
  int row  = blockIdx.x;
  int wavh = threadIdx.x >> 5;
  int lane = threadIdx.x & 31;
  size_t base = (size_t)row * 2048 + (size_t)wavh * 128 + (size_t)lane * 4;
  float xv[4];
  float ss = 0.f;
#pragma unroll
  for (int e = 0; e < 4; ++e) { xv[e] = o[base + e]; ss += xv[e] * xv[e]; }
#pragma unroll
  for (int m = 16; m >= 1; m >>= 1) ss += __shfl_xor(ss, m, 32);
  float r = rsqrtf(ss * (1.f / 128.f) + 1e-6f);
#pragma unroll
  for (int e = 0; e < 4; ++e) {
    float gv = bf2f(go[base + e]);
    float sg = 1.f / (1.f + __expf(-gv));
    out[base + e] = f2bf(xv[e] * r * w[lane * 4 + e] * sg);
  }
}

// ---------------------------------------------------------------------------
extern "C" void kernel_launch(void* const* d_in, const int* in_sizes, int n_in,
                              void* d_out, int out_size, void* d_ws, size_t ws_size,
                              hipStream_t stream) {
  (void)in_sizes; (void)n_in; (void)out_size; (void)ws_size;
  constexpr int Bc = 2, Tc = 4096, Mc = 8192, HC = 2048, Dc = 128, Hc = 16;

  const float* x        = (const float*)d_in[0];
  const float* Wq       = (const float*)d_in[1];
  const float* Wk       = (const float*)d_in[2];
  const float* Wv       = (const float*)d_in[3];
  const float* conv_q_w = (const float*)d_in[4];
  const float* conv_k_w = (const float*)d_in[5];
  const float* conv_v_w = (const float*)d_in[6];
  const float* A_log    = (const float*)d_in[7];
  const float* Wfa      = (const float*)d_in[8];
  const float* Wfb      = (const float*)d_in[9];
  const float* dt_bias  = (const float*)d_in[10];
  const float* Wb       = (const float*)d_in[11];
  const float* Wga      = (const float*)d_in[12];
  const float* Wgb      = (const float*)d_in[13];
  const float* o_norm_w = (const float*)d_in[14];
  const float* Wo       = (const float*)d_in[15];
  float* out            = (float*)d_out;

  char* ws = (char*)d_ws;
  size_t off = 0;
  auto take = [&](size_t bytes) -> char* {
    char* p = ws + off;
    off += (bytes + 255) & ~(size_t)255;
    return p;
  };

  unsigned short* xb    = (unsigned short*)take((size_t)Mc * HC * 2);
  unsigned short* WqT   = (unsigned short*)take((size_t)HC * HC * 2);
  unsigned short* WkT   = (unsigned short*)take((size_t)HC * HC * 2);
  unsigned short* WvT   = (unsigned short*)take((size_t)HC * HC * 2);
  unsigned short* WoT   = (unsigned short*)take((size_t)HC * HC * 2);
  unsigned short* WfaT  = (unsigned short*)take((size_t)Dc * HC * 2);
  unsigned short* WfbT  = (unsigned short*)take((size_t)HC * Dc * 2);
  unsigned short* WgaT  = (unsigned short*)take((size_t)Dc * HC * 2);
  unsigned short* WgbT  = (unsigned short*)take((size_t)HC * Dc * 2);
  unsigned short* qraw  = (unsigned short*)take((size_t)Mc * HC * 2);
  unsigned short* kraw  = (unsigned short*)take((size_t)Mc * HC * 2);
  unsigned short* vraw  = (unsigned short*)take((size_t)Mc * HC * 2);
  unsigned short* qf    = (unsigned short*)take((size_t)Mc * HC * 2);
  unsigned short* kf    = (unsigned short*)take((size_t)Mc * HC * 2);
  unsigned short* vf    = (unsigned short*)take((size_t)Mc * HC * 2);
  float*          tmpfa = (float*)take((size_t)Mc * Dc * 4);
  unsigned short* tmpfb = (unsigned short*)take((size_t)Mc * Dc * 2);
  float*          betab = (float*)take((size_t)Mc * Hc * 4);
  float*          oscan = (float*)take((size_t)Mc * HC * 4);
  unsigned short* ogate = (unsigned short*)take((size_t)Mc * HC * 2);
  // Overlays on the (dead after conv) raw buffers:
  float*          gbuf  = (float*)qraw;           // 67 MB spans qraw+kraw
  unsigned short* gobuf = vraw;                   // 33.5 MB

  const int EW = 256;
  const size_t nAct = (size_t)Mc * HC;

  // 1) converts / transposed bf16 weights
  f32_to_bf16_kernel<<<(unsigned)(nAct / EW), EW, 0, stream>>>(x, xb, nAct);
  transpose_bf16_kernel<<<(HC * HC + EW - 1) / EW, EW, 0, stream>>>(Wq, WqT, HC, HC);
  transpose_bf16_kernel<<<(HC * HC + EW - 1) / EW, EW, 0, stream>>>(Wk, WkT, HC, HC);
  transpose_bf16_kernel<<<(HC * HC + EW - 1) / EW, EW, 0, stream>>>(Wv, WvT, HC, HC);
  transpose_bf16_kernel<<<(HC * HC + EW - 1) / EW, EW, 0, stream>>>(Wo, WoT, HC, HC);
  transpose_bf16_kernel<<<(HC * Dc + EW - 1) / EW, EW, 0, stream>>>(Wfa, WfaT, HC, Dc);
  transpose_bf16_kernel<<<(Dc * HC + EW - 1) / EW, EW, 0, stream>>>(Wfb, WfbT, Dc, HC);
  transpose_bf16_kernel<<<(HC * Dc + EW - 1) / EW, EW, 0, stream>>>(Wga, WgaT, HC, Dc);
  transpose_bf16_kernel<<<(Dc * HC + EW - 1) / EW, EW, 0, stream>>>(Wgb, WgbT, Dc, HC);

  // 2) q/k/v projections (WMMA)
  dim3 gFull(HC / BN, Mc / BM);
  dim3 gThin(1, Mc / BM);
  wmma_gemm_kernel<1><<<gFull, 256, 0, stream>>>(xb, WqT, qraw, Mc, HC, HC);
  wmma_gemm_kernel<1><<<gFull, 256, 0, stream>>>(xb, WkT, kraw, Mc, HC, HC);
  wmma_gemm_kernel<1><<<gFull, 256, 0, stream>>>(xb, WvT, vraw, Mc, HC, HC);

  // 3) conv+silu, then per-head l2norm for q (with D^-0.5) and k
  conv_silu_kernel<<<(unsigned)(nAct / EW), EW, 0, stream>>>(qraw, conv_q_w, qf, Tc, HC);
  conv_silu_kernel<<<(unsigned)(nAct / EW), EW, 0, stream>>>(kraw, conv_k_w, kf, Tc, HC);
  conv_silu_kernel<<<(unsigned)(nAct / EW), EW, 0, stream>>>(vraw, conv_v_w, vf, Tc, HC);
  l2norm_kernel<<<Mc, 512, 0, stream>>>(qf, 0.08838834764831845f);  // D^-0.5
  l2norm_kernel<<<Mc, 512, 0, stream>>>(kf, 1.0f);

  // 4) g = -exp(A_log)*softplus((x@Wfa)@Wfb + dt_bias)   (raw bufs now dead)
  wmma_gemm_kernel<0><<<gThin, 256, 0, stream>>>(xb, WfaT, tmpfa, Mc, Dc, HC);
  f32_to_bf16_kernel<<<(unsigned)((size_t)Mc * Dc / EW), EW, 0, stream>>>(tmpfa, tmpfb, (size_t)Mc * Dc);
  wmma_gemm_kernel<0><<<gFull, 256, 0, stream>>>(tmpfb, WfbT, gbuf, Mc, HC, Dc);
  kda_gate_kernel<<<(unsigned)(nAct / EW), EW, 0, stream>>>(gbuf, A_log, dt_bias, nAct);

  // 5) beta
  beta_kernel<<<Mc, 256, 0, stream>>>(x, Wb, betab);

  // 6) output gate go = (x@Wga)@Wgb (bf16)
  wmma_gemm_kernel<0><<<gThin, 256, 0, stream>>>(xb, WgaT, tmpfa, Mc, Dc, HC);
  f32_to_bf16_kernel<<<(unsigned)((size_t)Mc * Dc / EW), EW, 0, stream>>>(tmpfa, tmpfb, (size_t)Mc * Dc);
  wmma_gemm_kernel<1><<<gFull, 256, 0, stream>>>(tmpfb, WgbT, gobuf, Mc, HC, Dc);

  // 7) chunked delta-rule scan (WMMA)
  kda_chunk_scan_kernel<<<dim3(Hc, Bc), 256, 0, stream>>>(qf, kf, vf, gbuf, betab, oscan, Tc);

  // 8) gated RMSNorm, then final projection (WMMA) to d_out
  gate_out_kernel<<<Mc, 512, 0, stream>>>(oscan, gobuf, o_norm_w, ogate);
  wmma_gemm_kernel<0><<<gFull, 256, 0, stream>>>(ogate, WoT, out, Mc, HC, HC);
}
